// g_sgcn_63660005261790
// MI455X (gfx1250) — matrix operand order repaired and checked by reading, not compiled, not run on hardware
//
#include <hip/hip_runtime.h>
#include <hip/hip_bf16.h>
#include <stdint.h>

#define BATCH   64
#define NODES   207
#define HIDDEN  64
#define NPAD    224     // node/K dim padded to multiple of 32
#define ROWSPAD 208     // node rows padded to multiple of 16
#define NTILES  13      // ceil(207/16)
#define STEPS   100
#define DT      0.01f
#define SQRT_DT 0.1f
#define YELEMS  (BATCH * NODES * HIDDEN)   // 847872
#define YBFELEMS (BATCH * HIDDEN * NPAD)   // 917504, transposed bf16 state
#define OUTCOLS (HIDDEN * HIDDEN)          // 4096
#define ZROWS   (BATCH * NODES)            // 13248 = 828*16

typedef __attribute__((ext_vector_type(16))) __bf16   v16bf;
typedef __attribute__((ext_vector_type(8)))  float    v8f;
typedef __attribute__((ext_vector_type(8)))  uint32_t v8u;

static __device__ inline unsigned short f2bf(float f) {
    uint32_t u = __float_as_uint(f);
    u += 0x7FFFu + ((u >> 16) & 1u);          // round-to-nearest-even
    return (unsigned short)(u >> 16);
}

static __device__ inline v16bf make_frag(uint4 q0, uint4 q1) {
    v8u p = { q0.x, q0.y, q0.z, q0.w, q1.x, q1.y, q1.z, q1.w };
    return __builtin_bit_cast(v16bf, p);
}

static __device__ inline v8f wmma_bf16(v16bf a, v16bf b, v8f c) {
    return __builtin_amdgcn_wmma_f32_16x16x32_bf16(
        false, a, false, b, (short)0, c, false, false);
}

// ---- deterministic per-(step, element) gaussian noise -----------------------
static __device__ inline uint32_t h32(uint32_t x) {
    x ^= x >> 16; x *= 0x7feb352du;
    x ^= x >> 15; x *= 0x846ca68bu;
    x ^= x >> 16; return x;
}
static __device__ inline float gnoise(uint32_t step, uint32_t idx) {
    uint32_t s  = h32(idx ^ (step * 0x9E3779B9u + 0x632BE5ABu));
    uint32_t u1 = h32(s ^ 0x85ebca6bu);
    uint32_t u2 = h32(s + 0xc2b2ae35u);
    float f1 = ((float)u1 + 1.0f) * 2.3283064365386963e-10f;  // (0,1]
    float f2 = (float)u2 * 2.3283064365386963e-10f;           // [0,1)
    return sqrtf(-2.0f * __logf(f1)) * __cosf(6.28318530718f * f2);
}

// ---- prep: A (row-major, padded) -> bf16 ------------------------------------
__global__ void k_prep_A(const float* __restrict__ A, unsigned short* __restrict__ Abf) {
    int i = blockIdx.x * blockDim.x + threadIdx.x;
    if (i >= ROWSPAD * NPAD) return;
    int r = i / NPAD, c = i % NPAD;
    float v = (r < NODES && c < NODES) ? A[r * NODES + c] : 0.0f;
    Abf[i] = f2bf(v);
}

// ---- prep: W[k][n] -> Wt[n][k] bf16 (K-contiguous for B fragments) ----------
__global__ void k_prep_WT(const float* __restrict__ W, unsigned short* __restrict__ Wt,
                          int K, int N) {
    int i = blockIdx.x * blockDim.x + threadIdx.x;
    if (i >= K * N) return;
    int n = i / K, k = i % K;
    Wt[i] = f2bf(W[k * N + n]);
}

// ---- init: transposed bf16 state from x; zero both buffers (pads stay 0) ----
__global__ void k_init_ybf(const float* __restrict__ x,
                           unsigned short* __restrict__ y0,
                           unsigned short* __restrict__ y1) {
    int i = blockIdx.x * blockDim.x + threadIdx.x;
    if (i >= YBFELEMS) return;
    int n  = i % NPAD;
    int bh = i / NPAD;
    int h  = bh % HIDDEN, b = bh / HIDDEN;
    unsigned short v = 0;
    if (n < NODES) v = f2bf(x[((size_t)b * NODES + n) * HIDDEN + h]);
    y0[i] = v;
    y1[i] = 0;
}

// ---- one Euler-Maruyama step ------------------------------------------------
// block = (node-tile, batch), 128 threads = 4 waves; wave wv owns hidden cols [wv*16, wv*16+16)
// Y^T bf16 state lives in global (L2-resident); B fragments load straight from it.
__global__ void __launch_bounds__(128)
k_sde_step(const float* __restrict__ yin, float* __restrict__ yout,
           const unsigned short* __restrict__ ybf_in,   // [b][h][NPAD] bf16
           unsigned short* __restrict__ ybf_out,        // [b][h][NPAD] bf16
           const unsigned short* __restrict__ Abf,
           const unsigned short* __restrict__ WtT,
           const unsigned short* __restrict__ WtD,
           const float* __restrict__ bT, const float* __restrict__ bD,
           int step)
{
    __shared__ __attribute__((aligned(16))) unsigned short Ml[16 * HIDDEN];   // [m][h] bf16

    const int tile = blockIdx.x;
    const int b    = blockIdx.y;
    const int tid  = threadIdx.x;
    const int lane = tid & 31;
    const int wv   = tid >> 5;
    const int nl   = lane & 15;   // row (A) / col (B,C,D) within tile
    const int kh   = lane >> 4;   // K-half selector

    // GEMM1: M[16 x 64] = A_tile[16 x 207] @ Y[207 x 64], K padded to 224
    v8f acc = {};
    const unsigned short* arow = Abf + ((size_t)tile * 16 + nl) * NPAD;
    const unsigned short* brow = ybf_in + ((size_t)b * HIDDEN + (wv * 16 + nl)) * NPAD;
    #pragma unroll
    for (int k0 = 0; k0 < NPAD; k0 += 32) {
        const unsigned short* ap = arow + k0 + kh * 8;        // A layout: 8+8 split
        v16bf afrag = make_frag(*(const uint4*)ap, *(const uint4*)(ap + 16));
        const unsigned short* bp = brow + k0 + kh * 8;        // B layout: 8+8 split
        v16bf bfrag = make_frag(*(const uint4*)bp, *(const uint4*)(bp + 16));
        acc = wmma_bf16(afrag, bfrag, acc);
    }

    // spill M tile to LDS as bf16 for use as A-matrix of GEMM2
    const int hcol = wv * 16 + nl;
    #pragma unroll
    for (int v = 0; v < 8; ++v) {
        int m = v + 8 * kh;                     // C/D layout: VGPR v, lane-half kh
        Ml[m * HIDDEN + hcol] = f2bf(acc[v]);
    }
    __syncthreads();

    // GEMM2: F_pre = M @ W_trend, G_pre = M @ W_diff  (K = 64)
    v8f accF = {}, accG = {};
    #pragma unroll
    for (int k0 = 0; k0 < HIDDEN; k0 += 32) {
        const unsigned short* mp = &Ml[nl * HIDDEN + k0 + kh * 8];
        v16bf mfrag = make_frag(*(const uint4*)mp, *(const uint4*)(mp + 16));
        const unsigned short* wt = WtT + (hcol)*HIDDEN + k0 + kh * 8;
        v16bf tfrag = make_frag(*(const uint4*)wt, *(const uint4*)(wt + 16));
        accF = wmma_bf16(mfrag, tfrag, accF);
        const unsigned short* wd = WtD + (hcol)*HIDDEN + k0 + kh * 8;
        v16bf dfrag = make_frag(*(const uint4*)wd, *(const uint4*)(wd + 16));
        accG = wmma_bf16(mfrag, dfrag, accG);
    }

    // y' = y + 0.1*tanh(F_pre+bT)*dt + 0.1*tanh(G_pre+bD)*sqrt(dt)*N(0,1)
    // also emit the transposed bf16 copy for the next step's B fragments
    const int   hout = hcol;
    const float bt = bT[hout], bd = bD[hout];
    #pragma unroll
    for (int v = 0; v < 8; ++v) {
        int node = tile * 16 + v + 8 * kh;
        if (node < NODES) {
            uint32_t idx = (uint32_t)b * (NODES * HIDDEN) + (uint32_t)node * HIDDEN + hout;
            float F = 0.1f * tanhf(accF[v] + bt);
            float G = 0.1f * tanhf(accG[v] + bd);
            float ynew = yin[idx] + F * DT + G * (SQRT_DT * gnoise((uint32_t)step, idx));
            yout[idx] = ynew;
            ybf_out[((size_t)b * HIDDEN + hout) * NPAD + node] = f2bf(ynew);
        }
    }
}

// ---- final projection: out = tanh(Z[13248,64] @ W_out[64,4096] + b_out) -----
__global__ void __launch_bounds__(128)
k_out(const float* __restrict__ z, const unsigned short* __restrict__ WtO,
      const float* __restrict__ bO, float* __restrict__ out)
{
    __shared__ __attribute__((aligned(16))) unsigned short Zl[16 * HIDDEN];
    const int mt  = blockIdx.x;           // 0..827
    const int nb  = blockIdx.y;           // 0..63
    const int tid = threadIdx.x;
    const int lane = tid & 31, wv = tid >> 5;
    const int nl = lane & 15, kh = lane >> 4;

    for (int i = tid; i < 16 * HIDDEN; i += 128)
        Zl[i] = f2bf(z[(size_t)mt * 16 * HIDDEN + i]);
    __syncthreads();

    const int n = (nb * 4 + wv) * 16 + nl;  // output column 0..4095
    v8f acc = {};
    #pragma unroll
    for (int k0 = 0; k0 < HIDDEN; k0 += 32) {
        const unsigned short* ap = &Zl[nl * HIDDEN + k0 + kh * 8];
        v16bf afrag = make_frag(*(const uint4*)ap, *(const uint4*)(ap + 16));
        const unsigned short* bp = WtO + (size_t)n * HIDDEN + k0 + kh * 8;
        v16bf bfrag = make_frag(*(const uint4*)bp, *(const uint4*)(bp + 16));
        acc = wmma_bf16(afrag, bfrag, acc);
    }
    const float bo = bO[n];
    #pragma unroll
    for (int v = 0; v < 8; ++v) {
        int m = mt * 16 + v + 8 * kh;
        out[(size_t)m * OUTCOLS + n] = tanhf(acc[v] + bo);
    }
}

extern "C" void kernel_launch(void* const* d_in, const int* in_sizes, int n_in,
                              void* d_out, int out_size, void* d_ws, size_t ws_size,
                              hipStream_t stream) {
    const float* x   = (const float*)d_in[0];
    const float* A   = (const float*)d_in[1];
    const float* W_t = (const float*)d_in[2];
    const float* b_t = (const float*)d_in[3];
    const float* W_d = (const float*)d_in[4];
    const float* b_d = (const float*)d_in[5];
    const float* W_o = (const float*)d_in[6];
    const float* b_o = (const float*)d_in[7];
    float* out = (float*)d_out;

    char* ws = (char*)d_ws;
    size_t off = 0;
    auto alloc = [&](size_t bytes) -> void* {
        void* p = ws + off;
        off += (bytes + 255) & ~(size_t)255;
        return p;
    };
    unsigned short* Abf  = (unsigned short*)alloc((size_t)ROWSPAD * NPAD * 2);
    unsigned short* WtT  = (unsigned short*)alloc((size_t)HIDDEN * HIDDEN * 2);
    unsigned short* WtD  = (unsigned short*)alloc((size_t)HIDDEN * HIDDEN * 2);
    unsigned short* WtO  = (unsigned short*)alloc((size_t)OUTCOLS * HIDDEN * 2);
    float* ybuf0 = (float*)alloc((size_t)YELEMS * 4);
    float* ybuf1 = (float*)alloc((size_t)YELEMS * 4);
    unsigned short* ybf0 = (unsigned short*)alloc((size_t)YBFELEMS * 2);
    unsigned short* ybf1 = (unsigned short*)alloc((size_t)YBFELEMS * 2);
    (void)ws_size; (void)in_sizes; (void)n_in; (void)out_size;

    // constant conversions + state init (cheap; once per launch, deterministic)
    k_prep_A  <<<(ROWSPAD * NPAD + 255) / 256, 256, 0, stream>>>(A, Abf);
    k_prep_WT <<<(HIDDEN * HIDDEN + 255) / 256, 256, 0, stream>>>(W_t, WtT, HIDDEN, HIDDEN);
    k_prep_WT <<<(HIDDEN * HIDDEN + 255) / 256, 256, 0, stream>>>(W_d, WtD, HIDDEN, HIDDEN);
    k_prep_WT <<<(HIDDEN * OUTCOLS + 255) / 256, 256, 0, stream>>>(W_o, WtO, HIDDEN, OUTCOLS);
    k_init_ybf<<<(YBFELEMS + 255) / 256, 256, 0, stream>>>(x, ybf0, ybf1);

    hipMemcpyAsync(ybuf0, x, (size_t)YELEMS * sizeof(float),
                   hipMemcpyDeviceToDevice, stream);

    dim3 gs(NTILES, BATCH), bs(128);
    for (int s = 0; s < STEPS; ++s) {
        const float*          yi  = (s & 1) ? ybuf1 : ybuf0;
        float*                yo  = (s & 1) ? ybuf0 : ybuf1;
        const unsigned short* ybi = (s & 1) ? ybf1  : ybf0;
        unsigned short*       ybo = (s & 1) ? ybf0  : ybf1;
        k_sde_step<<<gs, bs, 0, stream>>>(yi, yo, ybi, ybo, Abf, WtT, WtD, b_t, b_d, s);
    }
    // STEPS even -> final f32 state back in ybuf0
    dim3 go(ZROWS / 16, OUTCOLS / (16 * 4));
    k_out<<<go, 128, 0, stream>>>(ybuf0, WtO, b_o, out);
}
